// MoeMlp_84971632984083
// MI455X (gfx1250) — matrix operand order repaired and checked by reading, not compiled, not run on hardware
//
#include <hip/hip_runtime.h>

// MoE SwiGLU (Mixtral top-2) for gfx1250 / MI455X.
//  - token gather per expert; fp32 weights streamed once, cvt->bf16 into LDS,
//    register-pipelined so global loads overlap WMMA (raw dscnt-only barriers)
//  - v_wmma_f32_16x16x32_bf16 everywhere, f32 accum
//  - GEMM1 A-tile: global_load_async_to_lds_b128 (ASYNCcnt)
//  - GEMM2 A-tile: TDM tensor_load_to_lds (TENSORcnt), wave0-issued, TDM pads
//    LDS rows (16dw + 4dw pad = 80 B) to match the conflict-free layout
//  - sched_group_barrier pins DS-read/WMMA software pipeline (frag loads run
//    one 16x16 tile ahead of their consuming WMMA)
//  - deterministic: no float atomics; combine kernel sums the two expert slots

#define D_MODEL 1024
#define D_FF    4096
#define NEXP    8
#define NTOK    2048
#define LP      40          // padded LDS row stride in halves (80 B)

#define USE_ASYNC_LDS 1

typedef __attribute__((ext_vector_type(16))) __bf16        v16bf;
typedef __attribute__((ext_vector_type(8)))  float         v8f;
typedef __attribute__((ext_vector_type(4)))  unsigned int  u32x4;

union Frag16 { v16bf v; u32x4 q[2]; };

#if __has_builtin(__builtin_amdgcn_sched_group_barrier)
#define SGB(mask, size) __builtin_amdgcn_sched_group_barrier((mask), (size), 0)
#else
#define SGB(mask, size)
#endif
// sched_group_barrier masks: 8 = MFMA/WMMA, 256 = DS read

__device__ __forceinline__ Frag16 load_bfrag(const __bf16* base) {
  Frag16 f;
  f.q[0] = *(const u32x4*)base;
  f.q[1] = *(const u32x4*)(base + 16);
  return f;
}

__device__ __forceinline__ void async_lds_b128(void* lds_dst, const void* gsrc) {
#if USE_ASYNC_LDS
  unsigned           loff = (unsigned)(size_t)lds_dst;   // flat[31:0] == LDS offset
  unsigned long long ga   = (unsigned long long)(size_t)gsrc;
  asm volatile("global_load_async_to_lds_b128 %0, %1, off"
               :: "v"(loff), "v"(ga) : "memory");
#else
  *(u32x4*)lds_dst = *(const u32x4*)gsrc;
#endif
}

__device__ __forceinline__ void async_wait() {
#if USE_ASYNC_LDS
  asm volatile("s_wait_asynccnt 0x0" ::: "memory");
#endif
}

// Workgroup barrier that only drains DScnt (LDS producer->consumer); does NOT
// wait loadcnt, so in-flight global loads keep overlapping compute.
__device__ __forceinline__ void wg_barrier_lds() {
  asm volatile("s_wait_dscnt 0x0\n\t"
               "s_barrier_signal -1\n\t"
               "s_barrier_wait -1" ::: "memory");
}

// ----------------------------------------------------------- TDM (gfx1250)
#if __has_builtin(__builtin_amdgcn_tensor_load_to_lds)
#define HAVE_TDM 1
typedef __attribute__((ext_vector_type(4))) unsigned int tdm_u32x4;
typedef __attribute__((ext_vector_type(8))) int          tdm_i32x8;
typedef __attribute__((ext_vector_type(4))) int          tdm_i32x4;

// 2D tile of 16-bit elements, tile_w elems x tile_h rows, row stride in elems.
// LDS rows padded: pad_interval = 16 DWORDs (code 3), pad_amount = 4 DWORDs
// (code 3) -> 64B data + 16B pad = 80 B rows == LP halves.
__device__ __forceinline__ void tdm_load_tile_b16(unsigned lds_off, const void* gaddr,
                                                  unsigned tile_w, unsigned tile_h,
                                                  unsigned stride_elems) {
  unsigned long long ga = (unsigned long long)(size_t)gaddr;
  tdm_u32x4 g0;
  g0.x = 1u;                                              // count=1 (user D#)
  g0.y = lds_off;                                         // lds_addr
  g0.z = (unsigned)(ga & 0xFFFFFFFFu);                    // global_addr[31:0]
  g0.w = (unsigned)((ga >> 32) & 0x01FFFFFFu) | (2u << 30); // addr[56:32]|type=2
  tdm_i32x8 g1;
  g1[0] = (int)((1u << 16) |                              // data_size=1 (2B)
                (1u << 20) | (3u << 22) | (3u << 25));    // pad_en, int=16dw, amt=4dw
  g1[1] = (int)((tile_w & 0xFFFFu) << 16);                // tensor_dim0 lo
  g1[2] = (int)(((tile_w >> 16) & 0xFFFFu) | ((tile_h & 0xFFFFu) << 16)); // td0 hi|td1 lo
  g1[3] = (int)(((tile_h >> 16) & 0xFFFFu) | ((tile_w & 0xFFFFu) << 16)); // td1 hi|tile_dim0
  g1[4] = (int)(tile_h & 0xFFFFu);                        // tile_dim1 (tile_dim2=0)
  g1[5] = (int)stride_elems;                              // tensor_dim0_stride lo
  g1[6] = 0;                                              // stride hi | dim1_stride lo
  g1[7] = 0;
  tdm_i32x4 z4 = {0, 0, 0, 0};
  tdm_i32x8 z8 = {0, 0, 0, 0, 0, 0, 0, 0};
  // 6-arg form (clang-23 / therock-10.0 headers): (g0, g1, g2, g3, g4, cpol)
  __builtin_amdgcn_tensor_load_to_lds(g0, g1, z4, z4, z8, 0);
}
__device__ __forceinline__ void tdm_wait() {
#if __has_builtin(__builtin_amdgcn_s_wait_tensorcnt)
  __builtin_amdgcn_s_wait_tensorcnt(0);
#else
  asm volatile("s_wait_tensorcnt 0x0" ::: "memory");
#endif
}
#else
#define HAVE_TDM 0
#endif

// ---------------------------------------------------------------- zero counts
__global__ __launch_bounds__(32) void moe_zero(int* __restrict__ counts) {
  if (threadIdx.x < NEXP) counts[threadIdx.x] = 0;
}

// ---------------------------------------------------------------- router
__global__ __launch_bounds__(256)
void moe_router(const float* __restrict__ x, const float* __restrict__ Wr,
                const float* __restrict__ br,
                int* __restrict__ counts, int* __restrict__ idx,
                int* __restrict__ sel0, int* __restrict__ sel1,
                int* __restrict__ pos0, int* __restrict__ pos1,
                float* __restrict__ wt0, float* __restrict__ wt1,
                __bf16* __restrict__ xb)
{
  const int t    = blockIdx.x;
  const int tid  = threadIdx.x;
  const int wave = tid >> 5, lane = tid & 31;
  __shared__ float slog[NEXP];

  float p = 0.f;
  for (int d = lane; d < D_MODEL; d += 32)
    p += x[(size_t)t * D_MODEL + d] * Wr[d * NEXP + wave];
  for (int off = 16; off; off >>= 1) p += __shfl_xor(p, off);
  if (lane == 0) slog[wave] = p + br[wave];

  for (int i = tid; i < D_MODEL; i += 256)
    xb[(size_t)t * D_MODEL + i] = (__bf16)x[(size_t)t * D_MODEL + i];

  __syncthreads();
  if (tid == 0) {
    float m = slog[0];
    for (int e = 1; e < NEXP; ++e) m = fmaxf(m, slog[e]);
    float pr[NEXP], s = 0.f;
    for (int e = 0; e < NEXP; ++e) { pr[e] = __expf(slog[e] - m); s += pr[e]; }
    for (int e = 0; e < NEXP; ++e) pr[e] /= s;
    int e0 = 0;
    for (int e = 1; e < NEXP; ++e) if (pr[e] > pr[e0]) e0 = e;
    int e1 = (e0 == 0) ? 1 : 0;
    for (int e = 0; e < NEXP; ++e) if (e != e0 && pr[e] > pr[e1]) e1 = e;
    float a = pr[e0], b = pr[e1], rs = a + b; a /= rs; b /= rs;
    int p0 = atomicAdd(&counts[e0], 1); idx[e0 * NTOK + p0] = t;
    int p1 = atomicAdd(&counts[e1], 1); idx[e1 * NTOK + p1] = t;
    sel0[t] = e0; sel1[t] = e1; pos0[t] = p0; pos1[t] = p1;
    wt0[t] = a;  wt1[t] = b;
  }
}

// ---------------------------------------------------------------- GEMM1
__global__ __launch_bounds__(256)
void moe_gemm1(const __bf16* __restrict__ xb,
               const float* __restrict__ Wg, const float* __restrict__ bg,
               const float* __restrict__ Wu, const float* __restrict__ bu,
               const int* __restrict__ counts, const int* __restrict__ idx,
               __bf16* __restrict__ H)
{
  const int e   = blockIdx.z;
  const int n0  = blockIdx.x * 128;
  const int m0  = blockIdx.y * 128;
  const int cnt = counts[e];
  if (m0 >= cnt) return;

  __shared__ __align__(16) __bf16 As [128 * LP];
  __shared__ __align__(16) __bf16 Bgs[128 * LP];
  __shared__ __align__(16) __bf16 Bus[128 * LP];
  __shared__ int toks[128];

  const int tid = threadIdx.x;
  if (tid < 128) {
    int g = m0 + tid;
    toks[tid] = idx[e * NTOK + ((g < cnt) ? g : 0)];
  }
  __syncthreads();

  const int lane = tid & 31, wm = tid >> 5;
  const int half = lane >> 4, l16 = lane & 15;

  const int ar = tid >> 1, ac = (tid & 1) * 16;     // A staging: 2 threads/row
  const __bf16* asrc_base = xb + (size_t)toks[ar] * D_MODEL + ac;
  __bf16* adst = &As[ar * LP + ac];

  const int kr = tid >> 3, nc0 = (tid & 7) * 16;    // B staging: 16 floats/thread
  const size_t wbase = (size_t)e * D_MODEL * D_FF;
  const float4* gptr = (const float4*)(Wg + wbase + (size_t)kr * D_FF + n0 + nc0);
  const float4* uptr = (const float4*)(Wu + wbase + (size_t)kr * D_FF + n0 + nc0);
  const size_t krow16 = (size_t)(32 * D_FF) / 4;

  v8f cg[8] = {}; v8f cu[8] = {};
  float4 rg[4], ru[4];
  #pragma unroll
  for (int j = 0; j < 4; ++j) { rg[j] = gptr[j]; ru[j] = uptr[j]; }

  const __bf16* afr = &As[(wm * 16 + l16) * LP];
  const __bf16* bg0 = &Bgs[l16 * LP + half * 8];
  const __bf16* bu0 = &Bus[l16 * LP + half * 8];

  for (int k0 = 0; k0 < D_MODEL; k0 += 32) {
    wg_barrier_lds();                                 // prev reads done
    async_lds_b128(adst,     asrc_base + k0);
    async_lds_b128(adst + 8, asrc_base + k0 + 8);
    #pragma unroll
    for (int j = 0; j < 4; ++j) {
      int nb = nc0 + j * 4;
      Bgs[(nb+0)*LP + kr] = (__bf16)rg[j].x;  Bgs[(nb+1)*LP + kr] = (__bf16)rg[j].y;
      Bgs[(nb+2)*LP + kr] = (__bf16)rg[j].z;  Bgs[(nb+3)*LP + kr] = (__bf16)rg[j].w;
      Bus[(nb+0)*LP + kr] = (__bf16)ru[j].x;  Bus[(nb+1)*LP + kr] = (__bf16)ru[j].y;
      Bus[(nb+2)*LP + kr] = (__bf16)ru[j].z;  Bus[(nb+3)*LP + kr] = (__bf16)ru[j].w;
    }
    if (k0 + 32 < D_MODEL) {                          // overlap with WMMA phase
      const float4* g2 = gptr + ((size_t)(k0/32 + 1)) * krow16;
      const float4* u2 = uptr + ((size_t)(k0/32 + 1)) * krow16;
      #pragma unroll
      for (int j = 0; j < 4; ++j) { rg[j] = g2[j]; ru[j] = u2[j]; }
    }
    async_wait();
    wg_barrier_lds();                                 // tiles ready

    Frag16 a;
    a.q[0] = *(const u32x4*)(afr + half * 8);
    a.q[1] = *(const u32x4*)(afr + 16 + half * 8);

    Frag16 fg[2], fu[2];
    fg[0] = load_bfrag(bg0);
    fu[0] = load_bfrag(bu0);
    #pragma unroll
    for (int i = 0; i < 8; ++i) {
      const int cur = i & 1;
      if (i < 7) {                                   // prefetch next frags
        fg[cur ^ 1] = load_bfrag(bg0 + (i + 1) * 16 * LP);
        fu[cur ^ 1] = load_bfrag(bu0 + (i + 1) * 16 * LP);
      }
      cg[i] = __builtin_amdgcn_wmma_f32_16x16x32_bf16(false, a.v, false, fg[cur].v,
                                                      (short)0, cg[i], false, false);
      cu[i] = __builtin_amdgcn_wmma_f32_16x16x32_bf16(false, a.v, false, fu[cur].v,
                                                      (short)0, cu[i], false, false);
    }
    // Pipeline shape: 10 DS reads (A-frag + B tiles 0,1), then {2 WMMA, 4 DS}
    // so fragment loads stay one 16x16 tile ahead of their consuming WMMA.
    SGB(256, 10);
    SGB(8, 2); SGB(256, 4);
    SGB(8, 2); SGB(256, 4);
    SGB(8, 2); SGB(256, 4);
    SGB(8, 2); SGB(256, 4);
    SGB(8, 2); SGB(256, 4);
    SGB(8, 2); SGB(256, 4);
    SGB(8, 4);
  }

  const int rowb = wm * 16 + half * 8;
  #pragma unroll
  for (int i = 0; i < 8; ++i) {
    int f = n0 + i * 16 + l16;
    float bgv = bg[e * D_FF + f], buv = bu[e * D_FF + f];
    #pragma unroll
    for (int j = 0; j < 8; ++j) {
      int grow = m0 + rowb + j;
      if (grow < cnt) {
        float gv = cg[i][j] + bgv;
        float uv = cu[i][j] + buv;
        float hv = (gv / (1.f + __expf(-gv))) * uv;
        H[((size_t)e * NTOK + grow) * D_FF + f] = (__bf16)hv;
      }
    }
  }
}

// ---------------------------------------------------------------- GEMM2
__global__ __launch_bounds__(256)
void moe_gemm2(const __bf16* __restrict__ H,
               const float* __restrict__ Wd, const float* __restrict__ bd,
               const int* __restrict__ counts,
               float* __restrict__ Y)
{
  const int e   = blockIdx.z;
  const int n0  = blockIdx.x * 128;
  const int m0  = blockIdx.y * 128;
  const int cnt = counts[e];
  if (m0 >= cnt) return;

  __shared__ __align__(16) __bf16 As[128 * LP];
  __shared__ __align__(16) __bf16 Bs[128 * LP];

  const int tid  = threadIdx.x;
  const int lane = tid & 31, wm = tid >> 5;
  const int half = lane >> 4, l16 = lane & 15;

  const __bf16* hbase = H + ((size_t)e * NTOK + m0) * D_FF;
#if !HAVE_TDM
  const int ar = tid >> 1, ac = (tid & 1) * 16;
  const __bf16* asrc_base = hbase + (size_t)ar * D_FF + ac;
  __bf16* adst = &As[ar * LP + ac];
#endif

  const int kr = tid >> 3, nc0 = (tid & 7) * 16;
  const size_t wbase = (size_t)e * D_FF * D_MODEL;
  const float4* dptr = (const float4*)(Wd + wbase + (size_t)kr * D_MODEL + n0 + nc0);
  const size_t krow16 = (size_t)(32 * D_MODEL) / 4;

  v8f c[8] = {};
  float4 rd[4];
  #pragma unroll
  for (int j = 0; j < 4; ++j) rd[j] = dptr[j];

  const __bf16* afr = &As[(wm * 16 + l16) * LP];
  const __bf16* bb0 = &Bs[l16 * LP + half * 8];

  for (int k0 = 0; k0 < D_FF; k0 += 32) {
    wg_barrier_lds();
#if HAVE_TDM
    if (wm == 0)                         // one DMA moves the whole 128x32 tile
      tdm_load_tile_b16((unsigned)(size_t)&As[0], hbase + k0,
                        /*tile_w=*/32, /*tile_h=*/128, /*stride=*/D_FF);
#else
    async_lds_b128(adst,     asrc_base + k0);
    async_lds_b128(adst + 8, asrc_base + k0 + 8);
#endif
    #pragma unroll
    for (int j = 0; j < 4; ++j) {
      int nb = nc0 + j * 4;
      Bs[(nb+0)*LP + kr] = (__bf16)rd[j].x;  Bs[(nb+1)*LP + kr] = (__bf16)rd[j].y;
      Bs[(nb+2)*LP + kr] = (__bf16)rd[j].z;  Bs[(nb+3)*LP + kr] = (__bf16)rd[j].w;
    }
    if (k0 + 32 < D_FF) {
      const float4* d2 = dptr + ((size_t)(k0/32 + 1)) * krow16;
      #pragma unroll
      for (int j = 0; j < 4; ++j) rd[j] = d2[j];
    }
#if HAVE_TDM
    if (wm == 0) tdm_wait();
#else
    async_wait();
#endif
    wg_barrier_lds();

    Frag16 a;
    a.q[0] = *(const u32x4*)(afr + half * 8);
    a.q[1] = *(const u32x4*)(afr + 16 + half * 8);

    Frag16 fb[2];
    fb[0] = load_bfrag(bb0);
    #pragma unroll
    for (int i = 0; i < 8; ++i) {
      const int cur = i & 1;
      if (i < 7) fb[cur ^ 1] = load_bfrag(bb0 + (i + 1) * 16 * LP);
      c[i] = __builtin_amdgcn_wmma_f32_16x16x32_bf16(false, a.v, false, fb[cur].v,
                                                     (short)0, c[i], false, false);
    }
    // 6 DS reads (A-frag + tiles 0,1), then {1 WMMA, 2 DS} x6, final 2 WMMA.
    SGB(256, 6);
    SGB(8, 1); SGB(256, 2);
    SGB(8, 1); SGB(256, 2);
    SGB(8, 1); SGB(256, 2);
    SGB(8, 1); SGB(256, 2);
    SGB(8, 1); SGB(256, 2);
    SGB(8, 1); SGB(256, 2);
    SGB(8, 2);
  }

  const int rowb = wm * 16 + half * 8;
  #pragma unroll
  for (int i = 0; i < 8; ++i) {
    int dcol = n0 + i * 16 + l16;
    float bdv = bd[e * D_MODEL + dcol];
    #pragma unroll
    for (int j = 0; j < 8; ++j) {
      int grow = m0 + rowb + j;
      if (grow < cnt)
        Y[((size_t)e * NTOK + grow) * D_MODEL + dcol] = c[i][j] + bdv;
    }
  }
}

// ---------------------------------------------------------------- combine
__global__ __launch_bounds__(256)
void moe_combine(const float* __restrict__ Y,
                 const int* __restrict__ sel0, const int* __restrict__ sel1,
                 const int* __restrict__ pos0, const int* __restrict__ pos1,
                 const float* __restrict__ wt0, const float* __restrict__ wt1,
                 float* __restrict__ out)
{
  const int t = blockIdx.x;
  const float a = wt0[t], b = wt1[t];
  const float* y0 = Y + ((size_t)sel0[t] * NTOK + pos0[t]) * D_MODEL;
  const float* y1 = Y + ((size_t)sel1[t] * NTOK + pos1[t]) * D_MODEL;
  for (int i = threadIdx.x; i < D_MODEL; i += 256)
    out[(size_t)t * D_MODEL + i] = a * y0[i] + b * y1[i];
}

// ---------------------------------------------------------------- launch
extern "C" void kernel_launch(void* const* d_in, const int* in_sizes, int n_in,
                              void* d_out, int out_size, void* d_ws, size_t ws_size,
                              hipStream_t stream)
{
  const float* x  = (const float*)d_in[0];
  const float* Wr = (const float*)d_in[1];
  const float* br = (const float*)d_in[2];
  const float* Wg = (const float*)d_in[3];
  const float* bg = (const float*)d_in[4];
  const float* Wu = (const float*)d_in[5];
  const float* bu = (const float*)d_in[6];
  const float* Wd = (const float*)d_in[7];
  const float* bd = (const float*)d_in[8];
  float* out = (float*)d_out;

  char* ws = (char*)d_ws;
  size_t off = 0;
  auto alloc = [&](size_t bytes) -> char* {
    off = (off + 255) & ~(size_t)255;
    char* p = ws + off; off += bytes; return p;
  };
  int*    counts = (int*)   alloc(NEXP * sizeof(int));
  int*    idx    = (int*)   alloc((size_t)NEXP * NTOK * sizeof(int));
  int*    sel0   = (int*)   alloc(NTOK * sizeof(int));
  int*    sel1   = (int*)   alloc(NTOK * sizeof(int));
  int*    pos0   = (int*)   alloc(NTOK * sizeof(int));
  int*    pos1   = (int*)   alloc(NTOK * sizeof(int));
  float*  wt0    = (float*) alloc(NTOK * sizeof(float));
  float*  wt1    = (float*) alloc(NTOK * sizeof(float));
  __bf16* xb     = (__bf16*)alloc((size_t)NTOK * D_MODEL * 2);
  __bf16* H      = (__bf16*)alloc((size_t)NEXP * NTOK * D_FF * 2);      // 128 MiB
  float*  Y      = (float*) alloc((size_t)NEXP * NTOK * D_MODEL * 4);   // 64 MiB

  moe_zero<<<1, 32, 0, stream>>>(counts);
  moe_router<<<NTOK, 256, 0, stream>>>(x, Wr, br, counts, idx,
                                       sel0, sel1, pos0, pos1, wt0, wt1, xb);
  moe_gemm1<<<dim3(D_FF / 128, NTOK / 128, NEXP), 256, 0, stream>>>(
      xb, Wg, bg, Wu, bu, counts, idx, H);
  moe_gemm2<<<dim3(D_MODEL / 128, NTOK / 128, NEXP), 256, 0, stream>>>(
      H, Wd, bd, counts, Y);
  moe_combine<<<NTOK, 256, 0, stream>>>(Y, sel0, sel1, pos0, pos1, wt0, wt1, out);
}
